// LevelAttention_loss_10033043604171
// MI455X (gfx1250) — matrix operand order repaired and checked by reading, not compiled, not run on hardware
//
#include <hip/hip_runtime.h>
#include <hip/hip_bf16.h>
#include <math.h>

// CDNA5 WMMA vector types
typedef __attribute__((ext_vector_type(16))) _Float16 v16h;
typedef __attribute__((ext_vector_type(8)))  float    v8f;

#define BATCH 32
#define NBOX  64
#define EPSV  1e-7f

__device__ __forceinline__ v8f wmma_f32_16x16x32_f16(v16h a, v16h b, v8f c) {
#if defined(__gfx1250__)
  // 8 args: (neg_a, A, neg_b, B, c_mod, C, reuse_a, reuse_b)
  return __builtin_amdgcn_wmma_f32_16x16x32_f16(false, a, false, b, (short)0, c,
                                                false, false);
#else
  (void)a; (void)b;
  return c; // host-pass placeholder; device pass always takes the branch above
#endif
}

__global__ void zero_out_kernel(float* out) { out[0] = 0.0f; }

// One wave (32 threads) per 16x16 output tile of one image at one pyramid level.
// Rasterization mask = (y_in^T @ x_in) > 0 computed with two K=32 f16 WMMAs
// (counts <= 64 are exact in f16 operands / f32 accumulation).
template<int LID, int HH, int WW>
__global__ __launch_bounds__(32)
void level_loss_kernel(const float* __restrict__ att,
                       const float* __restrict__ bboxs,
                       const int*   __restrict__ Hp,
                       const int*   __restrict__ Wp,
                       float*       __restrict__ out)
{
  constexpr int TX = (WW + 15) / 16;
  const int tid  = threadIdx.x;          // 0..31 (wave32)
  const int b    = blockIdx.y;           // image
  const int tile = blockIdx.x;
  const int tx0  = (tile % TX) * 16;
  const int ty0  = (tile / TX) * 16;

  __shared__ float2 s_bx[NBOX];  // (ix1, ix2) per box, in att-pixel space
  __shared__ float2 s_by[NBOX];  // (iy1, iy2) per box; sentinel if not selected

  const float Wf = (float)Wp[0];
  const float Hf = (float)Hp[0];
  const float sx = (float)WW / Wf;
  const float sy = (float)HH / Hf;
  constexpr float base  = (float)(1 << (LID + 5));
  constexpr float min_a = base * base * 0.5f;
  constexpr float max_a = (base * 1.58f) * (base * 1.58f) * 2.0f;

  // Each lane preprocesses 2 of the 64 boxes.
  bool any_valid_l = false;
  #pragma unroll
  for (int j = 0; j < 2; ++j) {
    const int n = tid + 32 * j;
    const float* bb = bboxs + ((size_t)b * NBOX + n) * 5;
    const float x1 = bb[0], y1 = bb[1], x2 = bb[2], y2 = bb[3], lab = bb[4];
    const bool valid = (lab != -1.0f) && (x1 <= Wf) && (y1 <= Hf) &&
                       (x2 <= Wf) && (y2 <= Hf);
    any_valid_l |= valid;
    const float area = (x2 - x1) * (y2 - y1);
    const bool lv = valid && (area >= min_a) && (area <= max_a);
    const float ix1 = fmaxf(floorf(x1 * sx), 0.0f);
    const float iy1 = fmaxf(floorf(y1 * sy), 0.0f);
    const float ix2 = fminf(ceilf(x2 * sx) + 1.0f, (float)WW);
    const float iy2 = fminf(ceilf(y2 * sy) + 1.0f, (float)HH);
    s_bx[n] = make_float2(ix1, ix2);
    // Fold the level/validity mask into the y-range (empty range when off).
    s_by[n] = lv ? make_float2(iy1, iy2) : make_float2(3.0e9f, -3.0e9f);
  }
  const bool has_valid = (__ballot(any_valid_l) != 0ull);
  __syncthreads();

  const int half_id = tid >> 4;   // 0: lanes 0-15, 1: lanes 16-31
  const int lane16  = tid & 15;

  // ---- A fragments: y_in, 16 tile rows (M) x K boxes ----
  // 16-bit A 16x32 layout: lanes 0-15 hold K in {0..7,16..23},
  // lanes 16-31 hold K in {8..15,24..31}; M = lane & 15.
  const float yA = (float)(ty0 + lane16);
  v16h a0, a1;
  #pragma unroll
  for (int e = 0; e < 16; ++e) {
    const int k = e + (e & 8) + 8 * half_id;   // K index within [0,32)
    const float2 r0 = s_by[k];
    const float2 r1 = s_by[k + 32];
    a0[e] = (yA >= r0.x && yA < r0.y) ? (_Float16)1.0f : (_Float16)0.0f;
    a1[e] = (yA >= r1.x && yA < r1.y) ? (_Float16)1.0f : (_Float16)0.0f;
  }

  // ---- B fragments: x_in, K boxes x 16 tile cols (N) ----
  // 16-bit B 32x16 layout: lanes 0-15 hold K in {0..7,16..23}, lanes 16-31 hold K in {8..15,24..31};
  // N = lane & 15.
  const float xB = (float)(tx0 + lane16);
  v16h b0v, b1v;
  #pragma unroll
  for (int e = 0; e < 16; ++e) {
    const int k = e + (e & 8) + 8 * half_id;   // K index within [0,32)
    const float2 c0 = s_bx[k];
    const float2 c1 = s_bx[k + 32];
    b0v[e] = (xB >= c0.x && xB < c0.y) ? (_Float16)1.0f : (_Float16)0.0f;
    b1v[e] = (xB >= c1.x && xB < c1.y) ? (_Float16)1.0f : (_Float16)0.0f;
  }

  // counts[16x16] = y_in^T @ x_in over all 64 boxes (K=32 per WMMA).
  v8f c = {};
  c = wmma_f32_16x16x32_f16(a0, b0v, c);
  c = wmma_f32_16x16x32_f16(a1, b1v, c);

  // ---- BCE over this lane's 8 pixels ----
  // C layout: VGPR v -> row M = v + 8*half_id, col N = lane16.
  const int x = tx0 + lane16;
  float sum = 0.0f;
  if (x < WW) {
    const float* attb = att + (size_t)b * HH * WW;
    #pragma unroll
    for (int v = 0; v < 8; ++v) {
      const int y = ty0 + v + 8 * half_id;
      if (y < HH) {
        float p = attb[y * WW + x];
        p = fminf(fmaxf(p, EPSV), 1.0f - EPSV);
        const bool m = c[v] > 0.5f;            // count >= 1 -> covered
        sum += m ? -logf(p) : -log1pf(-p);
      }
    }
  }

  // wave32 reduction
  #pragma unroll
  for (int off = 16; off > 0; off >>= 1)
    sum += __shfl_down(sum, off, 32);

  if (tid == 0 && has_valid) {
    constexpr float scale = 1.0f / ((float)HH * (float)WW * 5.0f * (float)BATCH);
    atomicAdd(out, sum * scale);
  }
}

extern "C" void kernel_launch(void* const* d_in, const int* in_sizes, int n_in,
                              void* d_out, int out_size, void* d_ws, size_t ws_size,
                              hipStream_t stream) {
  (void)in_sizes; (void)n_in; (void)out_size; (void)d_ws; (void)ws_size;
  const float* att0  = (const float*)d_in[0];
  const float* att1  = (const float*)d_in[1];
  const float* att2  = (const float*)d_in[2];
  const float* att3  = (const float*)d_in[3];
  const float* att4  = (const float*)d_in[4];
  const float* bboxs = (const float*)d_in[5];
  const int*   Hp    = (const int*)d_in[6];
  const int*   Wp    = (const int*)d_in[7];
  float* out = (float*)d_out;

  zero_out_kernel<<<1, 1, 0, stream>>>(out);

  level_loss_kernel<0, 128, 128><<<dim3(64, BATCH), 32, 0, stream>>>(att0, bboxs, Hp, Wp, out);
  level_loss_kernel<1,  64,  64><<<dim3(16, BATCH), 32, 0, stream>>>(att1, bboxs, Hp, Wp, out);
  level_loss_kernel<2,  32,  32><<<dim3( 4, BATCH), 32, 0, stream>>>(att2, bboxs, Hp, Wp, out);
  level_loss_kernel<3,  16,  16><<<dim3( 1, BATCH), 32, 0, stream>>>(att3, bboxs, Hp, Wp, out);
  level_loss_kernel<4,   8,   8><<<dim3( 1, BATCH), 32, 0, stream>>>(att4, bboxs, Hp, Wp, out);
}
